// nTypeGAT_77627238908073
// MI455X (gfx1250) — matrix operand order repaired and checked by reading, not compiled, not run on hardware
//
#include <hip/hip_runtime.h>

typedef __bf16 bf16_t;
typedef bf16_t v8bf  __attribute__((ext_vector_type(8)));
typedef bf16_t v16bf __attribute__((ext_vector_type(16)));
typedef float  v8f   __attribute__((ext_vector_type(8)));
typedef float  v4f   __attribute__((ext_vector_type(4)));
typedef unsigned int uint32x4 __attribute__((ext_vector_type(4)));
typedef int    int32x8 __attribute__((ext_vector_type(8)));
typedef int    int32x4 __attribute__((ext_vector_type(4)));

#if defined(__has_builtin)
#if __has_builtin(__builtin_amdgcn_tensor_load_to_lds) && \
    __has_builtin(__builtin_amdgcn_s_wait_tensorcnt)
#define HAVE_TDM 1
#endif
#endif
#ifndef HAVE_TDM
#define HAVE_TDM 0
#endif

#define CB 8
#define CS 512
#define CD 256
#define CH 8
#define CDK 32
#define CDFF 1024
#define CDEP 40
#define CL 3
#define CM (CB * CS)
#define NEGV (-1.0e18f)

// ---- WMMA 16-bit fragment maps (ISA 7.12.2, wave32) ----
// A (MxK): lane = row m=l%16, half h=l/16; elems e -> K=16h+e (16 contiguous, same order as B)
// B (KxN) stored [N][K]-contiguous: lane = col n=l%16; elems e -> K=16h+e (16 contiguous)
// C/D: lane l reg r -> M = r + 8*(l/16), N = l%16

__device__ __forceinline__ v16bf cat8(v8bf lo, v8bf hi) {
  return __builtin_shufflevector(lo, hi, 0, 1, 2, 3, 4, 5, 6, 7, 8, 9, 10, 11,
                                 12, 13, 14, 15);
}

// =====================================================================
// elementwise f32 -> bf16 (+ optional f32 copy)
// =====================================================================
__global__ __launch_bounds__(256) void cvt_kernel(const float* __restrict__ in,
                                                  bf16_t* __restrict__ outb,
                                                  float* __restrict__ outf, int n) {
  int i = blockIdx.x * 256 + threadIdx.x;
  if (i < n) {
    float v = in[i];
    outb[i] = (bf16_t)v;
    if (outf) outf[i] = v;
  }
}

// f32 [L,K,N] -> bf16 [L,N,K] (weight pre-transpose, one-time)
__global__ __launch_bounds__(256) void cvtT_kernel(const float* __restrict__ in,
                                                   bf16_t* __restrict__ out,
                                                   int K, int N, int total) {
  int i = blockIdx.x * 256 + threadIdx.x;
  if (i >= total) return;
  int l = i / (K * N);
  int r = i - l * K * N;
  int k = r / N, n = r - k * N;
  out[(size_t)l * K * N + (size_t)n * K + k] = (bf16_t)in[i];
}

// =====================================================================
// GEMM: out = act(scale*(A[M,K]bf16 x W + bias)) (+residual)
// W pre-transposed bf16 [N,K]. 64x64 tile, BK=64, 8 waves x (16x32),
// double-buffered LDS so global loads overlap WMMA.
// =====================================================================
__global__ __launch_bounds__(256) void gemm_kernel(
    const bf16_t* __restrict__ A, const bf16_t* __restrict__ WT,
    const float* __restrict__ bias, const float* __restrict__ residual,
    float* __restrict__ outF, bf16_t* __restrict__ outB,
    int M, int N, int K, float scale, int relu) {
  __shared__ bf16_t As[2][64 * 72];  // [m][k], stride 72
  __shared__ bf16_t Bs[2][64 * 72];  // [n][k], stride 72
  int tid = threadIdx.x;
  int lane = tid & 31, wid = tid >> 5;
  int wm = wid & 3, wn = wid >> 2;
  int lm = lane & 15, lh = lane >> 4;
  int m0 = blockIdx.y * 64, n0 = blockIdx.x * 64;
  int r = tid >> 2, cg = (tid & 3) * 16;
  const bf16_t* aptr = A + (size_t)(m0 + r) * K + cg;
  const bf16_t* bptr = WT + (size_t)(n0 + r) * K + cg;

  {  // preload tile 0
    v8bf a0 = *(const v8bf*)aptr;
    v8bf a1 = *(const v8bf*)(aptr + 8);
    v8bf b0 = *(const v8bf*)bptr;
    v8bf b1 = *(const v8bf*)(bptr + 8);
    v8bf* da = (v8bf*)&As[0][r * 72 + cg];
    da[0] = a0; da[1] = a1;
    v8bf* db = (v8bf*)&Bs[0][r * 72 + cg];
    db[0] = b0; db[1] = b1;
  }
  __syncthreads();

  v8f acc0 = {}, acc1 = {};
  int nk = K >> 6;
  for (int it = 0; it < nk; ++it) {
    v8bf na0, na1, nb0, nb1;
    if (it + 1 < nk) {  // issue next tile's loads before computing
      const bf16_t* ap = aptr + (it + 1) * 64;
      const bf16_t* bp = bptr + (it + 1) * 64;
      na0 = *(const v8bf*)ap;
      na1 = *(const v8bf*)(ap + 8);
      nb0 = *(const v8bf*)bp;
      nb1 = *(const v8bf*)(bp + 8);
    }
    const bf16_t* Ac = As[it & 1];
    const bf16_t* Bc = Bs[it & 1];
#pragma unroll
    for (int ks = 0; ks < 2; ++ks) {
      const bf16_t* arow = &Ac[(wm * 16 + lm) * 72 + ks * 32];
      v16bf af = cat8(*(const v8bf*)(arow + 16 * lh),
                      *(const v8bf*)(arow + 16 * lh + 8));
      const bf16_t* b0r = &Bc[(wn * 32 + lm) * 72 + ks * 32 + 16 * lh];
      const bf16_t* b1r = &Bc[(wn * 32 + 16 + lm) * 72 + ks * 32 + 16 * lh];
      v16bf b0 = cat8(*(const v8bf*)b0r, *(const v8bf*)(b0r + 8));
      v16bf b1 = cat8(*(const v8bf*)b1r, *(const v8bf*)(b1r + 8));
      acc0 = __builtin_amdgcn_wmma_f32_16x16x32_bf16(false, af, false, b0,
                                                     (short)0, acc0, false, false);
      acc1 = __builtin_amdgcn_wmma_f32_16x16x32_bf16(false, af, false, b1,
                                                     (short)0, acc1, false, false);
    }
    if (it + 1 < nk) {
      v8bf* da = (v8bf*)&As[(it + 1) & 1][r * 72 + cg];
      da[0] = na0; da[1] = na1;
      v8bf* db = (v8bf*)&Bs[(it + 1) & 1][r * 72 + cg];
      db[0] = nb0; db[1] = nb1;
    }
    __syncthreads();
  }
#pragma unroll
  for (int rr = 0; rr < 8; ++rr) {
    int m = m0 + wm * 16 + rr + 8 * lh;
#pragma unroll
    for (int t = 0; t < 2; ++t) {
      int n = n0 + wn * 32 + t * 16 + lm;
      float v = (t ? acc1[rr] : acc0[rr]) + bias[n];
      v *= scale;
      if (relu) v = fmaxf(v, 0.f);
      if (residual) v += residual[(size_t)m * N + n];
      if (outF) outF[(size_t)m * N + n] = v;
      if (outB) outB[(size_t)m * N + n] = (bf16_t)v;
    }
  }
}

// =====================================================================
// LayerNorm over D=256, one block per row
// =====================================================================
__global__ __launch_bounds__(256) void ln_kernel(const float* __restrict__ x,
                                                 const float* __restrict__ g,
                                                 const float* __restrict__ b,
                                                 bf16_t* __restrict__ outb,
                                                 float* __restrict__ outf) {
  __shared__ float red[8];
  int row = blockIdx.x, t = threadIdx.x;
  float v = x[(size_t)row * CD + t];
  float s = v;
#pragma unroll
  for (int o = 16; o > 0; o >>= 1) s += __shfl_xor(s, o, 32);
  if ((t & 31) == 0) red[t >> 5] = s;
  __syncthreads();
  float tot = 0.f;
#pragma unroll
  for (int i = 0; i < 8; ++i) tot += red[i];
  float mean = tot * (1.f / CD);
  float d = v - mean;
  float sq = d * d;
  __syncthreads();
#pragma unroll
  for (int o = 16; o > 0; o >>= 1) sq += __shfl_xor(sq, o, 32);
  if ((t & 31) == 0) red[t >> 5] = sq;
  __syncthreads();
  float vtot = 0.f;
#pragma unroll
  for (int i = 0; i < 8; ++i) vtot += red[i];
  float outv = d * rsqrtf(vtot * (1.f / CD) + 1e-6f) * g[t] + b[t];
  if (outb) outb[(size_t)row * CD + t] = (bf16_t)outv;
  if (outf) outf[(size_t)row * CD + t] = outv;
}

#if HAVE_TDM
// Issue one TDM descriptor: 3D tile (x=40 dep, y=32 keys, z=16 queries), f32.
__device__ __forceinline__ void tdm_load_struct(const float* gsrc, float* ldst) {
  uint64_t gaddr = (uint64_t)(uintptr_t)gsrc;
  unsigned int laddr = (unsigned int)(uintptr_t)ldst;
  uint32x4 g0;
  g0[0] = 1u;  // count=1, user mode, no gather
  g0[1] = laddr;
  g0[2] = (unsigned int)(gaddr & 0xFFFFFFFFu);
  g0[3] = (unsigned int)((gaddr >> 32) & 0x01FFFFFFu) | 0x80000000u;  // type=2
  int32x8 g1;
  g1[0] = 0x00020000;               // data_size=4B
  g1[1] = (int)(CDEP << 16);        // tensor_dim0[15:0] in [31:16]
  g1[2] = (int)(CS << 16);          // tensor_dim0 hi=0 | tensor_dim1 lo=512
  g1[3] = (int)(CDEP << 16);        // tensor_dim1 hi=0 | tile_dim0=40
  g1[4] = (int)(32u | (16u << 16)); // tile_dim1=32, tile_dim2=16
  g1[5] = CDEP;                     // tensor_dim0_stride = 40
  g1[6] = (int)((unsigned)(CS * CDEP) << 16);  // dim1_stride lo16 (20480)
  g1[7] = 0;
  int32x4 g2;
  g2[0] = CS;              // tensor_dim2
  g2[1] = 0;               // tensor_dim3 (tile_dim3=0 -> unused)
  g2[2] = CS * CS * CDEP;  // tensor_dim2_stride = 10485760
  g2[3] = 0;               // stride hi | tile_dim3=0
  int32x4 g3;
  g3[0] = 0; g3[1] = 0; g3[2] = 0; g3[3] = 0;
  int32x8 g4;  // 6-arg toolchain form: extra group, unused
  g4[0] = 0; g4[1] = 0; g4[2] = 0; g4[3] = 0;
  g4[4] = 0; g4[5] = 0; g4[6] = 0; g4[7] = 0;
  __builtin_amdgcn_tensor_load_to_lds(g0, g1, g2, g3, g4, 0);
}
#endif

// =====================================================================
// Fused structure-aware attention. Block = (b, 16 queries); 8 waves = 8 heads.
// Structure tiles (512x40 f32, 80KB) double-buffered and streamed by the
// Tensor Data Mover one tile ahead of compute; WMMA projection to sk/sv;
// flash-style online softmax; WMMA p.v with transposed V.
// =====================================================================
__global__ __launch_bounds__(256) void attn_kernel(
    const bf16_t* __restrict__ qg, const bf16_t* __restrict__ kg,
    const bf16_t* __restrict__ vg, const float* __restrict__ structure,
    const int* __restrict__ mask, const int* __restrict__ kpm,
    const bf16_t* __restrict__ WskT, const float* __restrict__ bsk,
    const bf16_t* __restrict__ WsvT, const float* __restrict__ bsv,
    bf16_t* __restrict__ ctx, float alpha, float beta) {
#if HAVE_TDM
  __shared__ float sStF[2][512 * 40];  // structure tiles, raw f32 (TDM targets)
#else
  __shared__ bf16_t sStB[512 * 40];    // structure tile, bf16
#endif
  __shared__ bf16_t sQ[16 * 264];      // q tile (pre-scaled)
  __shared__ bf16_t sK[32 * 256];      // k tile, all heads
  __shared__ bf16_t sVT[256 * 40];     // v tile transposed [d][kk]
  __shared__ bf16_t sWkT[32 * 72];     // Wsk^T [n][k], zero-padded k 40..63
  __shared__ bf16_t sWvT[32 * 72];
  __shared__ bf16_t sSK[512 * 32];     // sk tile [row=(qm*32+kk)][d]
  __shared__ bf16_t sSVT[32 * 520];    // sv tile transposed [d][row]
  __shared__ bf16_t sP_arr[8 * 640];   // per-wave 16x40 p tiles

  int tid = threadIdx.x;
  int lane = tid & 31, wid = tid >> 5;
  int lm = lane & 15, lh = lane >> 4;
  int b = blockIdx.y;
  int q0 = blockIdx.x * 16;
  int h = wid;  // one wave per head
  const float* stbase = structure + ((size_t)b * CS + q0) * CS * CDEP;

  {  // q tile: 2 x b128 per thread
    int r = tid >> 4, cg = (tid & 15) * 16;
    const v8bf* src = (const v8bf*)(qg + ((size_t)b * CS + q0 + r) * CD + cg);
    v8bf* dst = (v8bf*)&sQ[r * 264 + cg];
    dst[0] = src[0];
    dst[1] = src[1];
  }
  for (int i = tid; i < 32 * 72; i += 256) {  // projection weights, K-padded
    int n = i / 72, k = i - n * 72;
    sWkT[i] = (k < CDEP) ? WskT[n * CDEP + k] : (bf16_t)0.f;
    sWvT[i] = (k < CDEP) ? WsvT[n * CDEP + k] : (bf16_t)0.f;
  }
#if HAVE_TDM
  if (wid == 0) tdm_load_struct(stbase, sStF[0]);  // prologue: tile 0 in flight
#endif
  __syncthreads();

  // q A-frag for this head (two contiguous 8-half chunks)
  const bf16_t* qrowf = &sQ[lm * 264 + h * 32];
  v16bf qf = cat8(*(const v8bf*)(qrowf + 16 * lh),
                  *(const v8bf*)(qrowf + 16 * lh + 8));

  v8f acc0 = {}, acc1 = {};
  float mmax[8], lsum[8];
#pragma unroll
  for (int r = 0; r < 8; ++r) { mmax[r] = -3.0e38f; lsum[r] = 0.f; }

  bf16_t* sP = sP_arr + wid * 640;

  for (int kt = 0; kt < CS / 32; ++kt) {
    int k0 = kt * 32;
    __syncthreads();  // prior head-phase done; alt structure buffer free
#if HAVE_TDM
    if (wid == 0) {  // stream tile kt+1 while tile kt is consumed
      if (kt + 1 < CS / 32) {
        tdm_load_struct(stbase + (size_t)(k0 + 32) * CDEP, sStF[(kt + 1) & 1]);
        __builtin_amdgcn_s_wait_tensorcnt(1);  // in-order: tile kt complete
      } else {
        __builtin_amdgcn_s_wait_tensorcnt(0);
      }
    }
#else
    {  // structure tile: 2 rows x 40 f32 -> bf16 per thread
#pragma unroll
      for (int rep = 0; rep < 2; ++rep) {
        int rr = tid * 2 + rep;
        int qm = rr >> 5, kk = rr & 31;
        const float* src =
            structure + (((size_t)b * CS + q0 + qm) * CS + k0 + kk) * CDEP;
        bf16_t* dst = &sStB[rr * 40];
#pragma unroll
        for (int j = 0; j < 40; ++j) dst[j] = (bf16_t)src[j];
      }
    }
#endif
    {  // k tile (row-major) + v tile (transposed [d][kk])
      int r = tid >> 3, cg = (tid & 7) * 32;
      const v8bf* ksrc = (const v8bf*)(kg + ((size_t)b * CS + k0 + r) * CD + cg);
      v8bf* kdst = (v8bf*)&sK[r * 256 + cg];
      const v8bf* vsrc = (const v8bf*)(vg + ((size_t)b * CS + k0 + r) * CD + cg);
#pragma unroll
      for (int j = 0; j < 4; ++j) {
        kdst[j] = ksrc[j];
        v8bf tv = vsrc[j];
#pragma unroll
        for (int e = 0; e < 8; ++e) sVT[(cg + j * 8 + e) * 40 + r] = tv[e];
      }
      if (kt + 1 < CS / 32) {  // prefetch next tiles toward L2
        __builtin_prefetch(kg + ((size_t)b * CS + k0 + 32 + r) * CD + cg, 0, 1);
        __builtin_prefetch(vg + ((size_t)b * CS + k0 + 32 + r) * CD + cg, 0, 1);
      }
    }
    __syncthreads();

    // ---- shared projection: sk/sv = struct(512x40) @ W(40x32) + bias ----
    for (int t = wid; t < 128; t += 8) {
      int p = t >> 6;
      int tt = t & 63;
      int mt = tt >> 1, nt = tt & 1;
      const bf16_t* wl = p ? sWvT : sWkT;
      int row = mt * 16 + lm;
      v8f pacc = {};
#pragma unroll
      for (int ks = 0; ks < 2; ++ks) {
        v16bf af;
#pragma unroll
        for (int c = 0; c < 2; ++c) {
          int kb = ks * 32 + 16 * lh + c * 8;
          if (kb < CDEP) {
#if HAVE_TDM
            v4f x0 = *(const v4f*)&sStF[kt & 1][row * 40 + kb];
            v4f x1 = *(const v4f*)&sStF[kt & 1][row * 40 + kb + 4];
#pragma unroll
            for (int j = 0; j < 4; ++j) {
              af[c * 8 + j] = (bf16_t)x0[j];
              af[c * 8 + 4 + j] = (bf16_t)x1[j];
            }
#else
            v8bf x = *(const v8bf*)&sStB[row * 40 + kb];
#pragma unroll
            for (int j = 0; j < 8; ++j) af[c * 8 + j] = x[j];
#endif
          } else {
#pragma unroll
            for (int j = 0; j < 8; ++j) af[c * 8 + j] = (bf16_t)0.f;
          }
        }
        const bf16_t* wr = &wl[(nt * 16 + lm) * 72 + ks * 32 + 16 * lh];
        v16bf bw = cat8(*(const v8bf*)wr, *(const v8bf*)(wr + 8));
        pacc = __builtin_amdgcn_wmma_f32_16x16x32_bf16(
            false, af, false, bw, (short)0, pacc, false, false);
      }
      int n = nt * 16 + lm;
      float bvv = p ? bsv[n] : bsk[n];
#pragma unroll
      for (int r = 0; r < 8; ++r) {
        int rr = mt * 16 + r + 8 * lh;
        float v = pacc[r] + bvv;
        if (p == 0) sSK[rr * 32 + n] = (bf16_t)v;
        else        sSVT[n * 520 + rr] = (bf16_t)v;
      }
    }
    __syncthreads();

    // ---- per-head: scores = q.k^T + alpha*q.sk ----
    v8f sc0, sc1;
    {
      v16bf bk0 = *(const v16bf*)&sK[lm * 256 + h * 32 + (lh << 4)];
      v16bf bk1 = *(const v16bf*)&sK[(16 + lm) * 256 + h * 32 + (lh << 4)];
      v8f z = {};
      sc0 = __builtin_amdgcn_wmma_f32_16x16x32_bf16(false, qf, false, bk0,
                                                    (short)0, z, false, false);
      sc1 = __builtin_amdgcn_wmma_f32_16x16x32_bf16(false, qf, false, bk1,
                                                    (short)0, z, false, false);
    }
    float s2[2][8];
#pragma unroll
    for (int ct = 0; ct < 2; ++ct) {
#pragma unroll
      for (int r = 0; r < 8; ++r) {
        int m = r + 8 * lh;
        int kk = ct * 16 + lm;
        const v8bf* qrow = (const v8bf*)&sQ[m * 264 + h * 32];
        const v8bf* krow = (const v8bf*)&sSK[(m * 32 + kk) * 32];
        float a = 0.f;
#pragma unroll
        for (int cgi = 0; cgi < 4; ++cgi) {
          v8bf qa = qrow[cgi], ka = krow[cgi];
#pragma unroll
          for (int j = 0; j < 8; ++j) a += (float)qa[j] * (float)ka[j];
        }
        s2[ct][r] = a;
      }
    }
    // ---- masks + online softmax ----
#pragma unroll
    for (int r = 0; r < 8; ++r) {
      int m = r + 8 * lh;
      float v0 = sc0[r] + alpha * s2[0][r];
      float v1 = sc1[r] + alpha * s2[1][r];
      size_t mrow = ((size_t)(b * CS + q0 + m)) * CS + k0;
      if (mask[mrow + lm] == 0 || kpm[b * CS + k0 + lm] == 0) v0 = NEGV;
      if (mask[mrow + 16 + lm] == 0 || kpm[b * CS + k0 + 16 + lm] == 0) v1 = NEGV;
      float t = fmaxf(v0, v1);
#pragma unroll
      for (int o = 8; o > 0; o >>= 1) t = fmaxf(t, __shfl_xor(t, o, 16));
      float mnew = fmaxf(mmax[r], t);
      float corr = __expf(mmax[r] - mnew);
      float p0 = __expf(v0 - mnew), p1 = __expf(v1 - mnew);
      float rsum = p0 + p1;
#pragma unroll
      for (int o = 8; o > 0; o >>= 1) rsum += __shfl_xor(rsum, o, 16);
      lsum[r] = lsum[r] * corr + rsum;
      mmax[r] = mnew;
      acc0[r] *= corr;
      acc1[r] *= corr;
      sP[m * 40 + lm] = (bf16_t)p0;
      sP[m * 40 + 16 + lm] = (bf16_t)p1;
    }
    asm volatile("" ::: "memory");  // keep same-wave LDS p writes before reads

    // ---- ctx += p.v (WMMA, V transposed in LDS) + beta * p.sv (VALU) ----
    const bf16_t* prowf = &sP[lm * 40];
    v16bf pf = cat8(*(const v8bf*)(prowf + 16 * lh),
                    *(const v8bf*)(prowf + 16 * lh + 8));
    const bf16_t* v0r = &sVT[(h * 32 + lm) * 40 + 16 * lh];
    const bf16_t* v1r = &sVT[(h * 32 + 16 + lm) * 40 + 16 * lh];
    v16bf bv0 = cat8(*(const v8bf*)v0r, *(const v8bf*)(v0r + 8));
    v16bf bv1 = cat8(*(const v8bf*)v1r, *(const v8bf*)(v1r + 8));
    acc0 = __builtin_amdgcn_wmma_f32_16x16x32_bf16(false, pf, false, bv0,
                                                   (short)0, acc0, false, false);
    acc1 = __builtin_amdgcn_wmma_f32_16x16x32_bf16(false, pf, false, bv1,
                                                   (short)0, acc1, false, false);
#pragma unroll
    for (int r = 0; r < 8; ++r) {
      int m = r + 8 * lh;
      const v8bf* prow = (const v8bf*)&sP[m * 40];
      const v8bf* s0 = (const v8bf*)&sSVT[lm * 520 + m * 32];
      const v8bf* s1 = (const v8bf*)&sSVT[(16 + lm) * 520 + m * 32];
      float a0 = 0.f, a1 = 0.f;
#pragma unroll
      for (int cgi = 0; cgi < 4; ++cgi) {
        v8bf pv = prow[cgi], x0 = s0[cgi], x1 = s1[cgi];
#pragma unroll
        for (int j = 0; j < 8; ++j) {
          float pw = (float)pv[j];
          a0 += pw * (float)x0[j];
          a1 += pw * (float)x1[j];
        }
      }
      acc0[r] += beta * a0;
      acc1[r] += beta * a1;
    }
  }

  // ---- finalize: ctx = acc / lsum ----
#pragma unroll
  for (int r = 0; r < 8; ++r) {
    int m = r + 8 * lh;
    float inv = 1.f / lsum[r];
    size_t base = ((size_t)b * CS + q0 + m) * CD + h * 32;
    ctx[base + lm] = (bf16_t)(acc0[r] * inv);
    ctx[base + 16 + lm] = (bf16_t)(acc1[r] * inv);
  }
}

// =====================================================================
// Host launcher
// =====================================================================
extern "C" void kernel_launch(void* const* d_in, const int* in_sizes, int n_in,
                              void* d_out, int out_size, void* d_ws,
                              size_t ws_size, hipStream_t stream) {
  (void)in_sizes; (void)n_in; (void)out_size; (void)ws_size;
  const float* src       = (const float*)d_in[0];
  const int*   mask      = (const int*)d_in[1];
  const int*   kpm       = (const int*)d_in[2];
  const float* structure = (const float*)d_in[3];
  const float* Wq = (const float*)d_in[4];   const float* bq = (const float*)d_in[5];
  const float* Wk = (const float*)d_in[6];   const float* bk = (const float*)d_in[7];
  const float* Wv = (const float*)d_in[8];   const float* bv = (const float*)d_in[9];
  const float* Wo = (const float*)d_in[10];  const float* bo = (const float*)d_in[11];
  const float* Wsk = (const float*)d_in[12]; const float* bsk = (const float*)d_in[13];
  const float* Wsv = (const float*)d_in[14]; const float* bsv = (const float*)d_in[15];
  const float* ln1g = (const float*)d_in[16]; const float* ln1b = (const float*)d_in[17];
  const float* Wf1 = (const float*)d_in[18]; const float* bf1 = (const float*)d_in[19];
  const float* Wf2 = (const float*)d_in[20]; const float* bf2 = (const float*)d_in[21];
  const float* lnfg = (const float*)d_in[22]; const float* lnfb = (const float*)d_in[23];

  char* ws = (char*)d_ws;
  bf16_t* xb   = (bf16_t*)ws;  ws += (size_t)CM * CD * 2;
  float*  xf   = (float*)ws;   ws += (size_t)CM * CD * 4;
  bf16_t* qb   = (bf16_t*)ws;  ws += (size_t)CM * CD * 2;
  bf16_t* kb   = (bf16_t*)ws;  ws += (size_t)CM * CD * 2;
  bf16_t* vb   = (bf16_t*)ws;  ws += (size_t)CM * CD * 2;
  bf16_t* ctxb = (bf16_t*)ws;  ws += (size_t)CM * CD * 2;
  float*  aof  = (float*)ws;   ws += (size_t)CM * CD * 4;
  bf16_t* hb   = (bf16_t*)ws;  ws += (size_t)CM * CD * 2;
  bf16_t* f1b  = (bf16_t*)ws;  ws += (size_t)CM * CDFF * 2;
  bf16_t* WqT  = (bf16_t*)ws;  ws += (size_t)CL * CD * CD * 2;
  bf16_t* WkT  = (bf16_t*)ws;  ws += (size_t)CL * CD * CD * 2;
  bf16_t* WvT  = (bf16_t*)ws;  ws += (size_t)CL * CD * CD * 2;
  bf16_t* WoT  = (bf16_t*)ws;  ws += (size_t)CL * CD * CD * 2;
  bf16_t* Wf1T = (bf16_t*)ws;  ws += (size_t)CL * CD * CDFF * 2;
  bf16_t* Wf2T = (bf16_t*)ws;  ws += (size_t)CL * CD * CDFF * 2;
  bf16_t* WskT = (bf16_t*)ws;  ws += (size_t)CL * CDEP * CDK * 2;
  bf16_t* WsvT = (bf16_t*)ws;  ws += (size_t)CL * CDEP * CDK * 2;

  auto cvtT = [&](const float* in, bf16_t* out, int K_, int N_) {
    int total = CL * K_ * N_;
    cvtT_kernel<<<(total + 255) / 256, 256, 0, stream>>>(in, out, K_, N_, total);
  };
  cvtT(Wq, WqT, CD, CD);
  cvtT(Wk, WkT, CD, CD);
  cvtT(Wv, WvT, CD, CD);
  cvtT(Wo, WoT, CD, CD);
  cvtT(Wf1, Wf1T, CD, CDFF);
  cvtT(Wf2, Wf2T, CDFF, CD);
  cvtT(Wsk, WskT, CDEP, CDK);
  cvtT(Wsv, WsvT, CDEP, CDK);

  auto gemm = [&](const bf16_t* A, const bf16_t* WT, const float* bias,
                  const float* resid, float* oF, bf16_t* oB, int M_, int N_,
                  int K_, float scale, int relu) {
    gemm_kernel<<<dim3(N_ / 64, M_ / 64), 256, 0, stream>>>(
        A, WT, bias, resid, oF, oB, M_, N_, K_, scale, relu);
  };

  const int n_act = CM * CD;
  cvt_kernel<<<(n_act + 255) / 256, 256, 0, stream>>>(src, xb, xf, n_act);

  const float qscale = 0.17677669529663687f;  // 1/sqrt(DK)
  for (int l = 0; l < CL; ++l) {
    size_t wdd = (size_t)l * CD * CD, wdf = (size_t)l * CD * CDFF;
    const float* bq_l = bq + (size_t)l * CD;
    const float* bk_l = bk + (size_t)l * CD;
    const float* bv_l = bv + (size_t)l * CD;
    const float* bo_l = bo + (size_t)l * CD;
    const float* bsk_l = bsk + (size_t)l * CDK;
    const float* bsv_l = bsv + (size_t)l * CDK;
    const float* g1 = ln1g + (size_t)l * CD;
    const float* b1 = ln1b + (size_t)l * CD;
    const float* bf1_l = bf1 + (size_t)l * CDFF;
    const float* bf2_l = bf2 + (size_t)l * CD;

    gemm(xb, WqT + wdd, bq_l, nullptr, nullptr, qb, CM, CD, CD, qscale, 0);
    gemm(xb, WkT + wdd, bk_l, nullptr, nullptr, kb, CM, CD, CD, 1.0f, 0);
    gemm(xb, WvT + wdd, bv_l, nullptr, nullptr, vb, CM, CD, CD, 1.0f, 0);

    attn_kernel<<<dim3(CS / 16, CB), 256, 0, stream>>>(
        qb, kb, vb, structure, mask, kpm, WskT + (size_t)l * CDEP * CDK, bsk_l,
        WsvT + (size_t)l * CDEP * CDK, bsv_l, ctxb, 1.0f, 1.0f);

    gemm(ctxb, WoT + wdd, bo_l, xf, aof, nullptr, CM, CD, CD, 1.0f, 0);
    ln_kernel<<<CM, CD, 0, stream>>>(aof, g1, b1, hb, nullptr);
    gemm(hb, Wf1T + wdf, bf1_l, nullptr, nullptr, f1b, CM, CDFF, CD, 1.0f, 1);
    gemm(f1b, Wf2T + wdf, bf2_l, aof, xf, xb, CM, CD, CDFF, 1.0f, 0);
  }
  ln_kernel<<<CM, CD, 0, stream>>>(xf, lnfg, lnfb, nullptr, (float*)d_out);
}
